// TALoss_23416161698369
// MI455X (gfx1250) — compile-verified, matrix operand order checked by reading, not run on hardware
//
#include <hip/hip_runtime.h>
#include <hip/hip_bf16.h>
#include <math.h>

// ---- static config (mirrors reference) ----
#define BB 32
#define MM 20
#define NCLS 80
#define REGM 16
#define NO 144
#define AA 8400          // 6400 + 1600 + 400
#define TOPKK 10
#define NBLK 1050        // (BB*AA)/256 exactly

typedef __attribute__((ext_vector_type(16))) _Float16 v16h;
typedef __attribute__((ext_vector_type(8)))  float    v8f;

// -------------------- helpers --------------------
__device__ __forceinline__ void anchor_info(int a, float& ax, float& ay, float& st) {
    if (a < 6400)      { st = 8.f;  ax = (float)(a % 80) + 0.5f;        ay = (float)(a / 80) + 0.5f; }
    else if (a < 8000) { int l = a - 6400; st = 16.f; ax = (float)(l % 40) + 0.5f; ay = (float)(l / 40) + 0.5f; }
    else               { int l = a - 8000; st = 32.f; ax = (float)(l % 20) + 0.5f; ay = (float)(l / 20) + 0.5f; }
}

__device__ __forceinline__ float feat_at(const float* f0, const float* f1, const float* f2,
                                         int b, int ch, int a) {
    if (a < 6400) return f0[((size_t)b * NO + ch) * 6400 + a];
    if (a < 8000) return f1[((size_t)b * NO + ch) * 1600 + (a - 6400)];
    return f2[((size_t)b * NO + ch) * 400 + (a - 8000)];
}

__device__ __forceinline__ float ciou_f(float b1x1, float b1y1, float b1x2, float b1y2,
                                        float b2x1, float b2y1, float b2x2, float b2y2) {
    const float eps = 1e-7f;
    float w1 = b1x2 - b1x1, h1 = b1y2 - b1y1 + eps;
    float w2 = b2x2 - b2x1, h2 = b2y2 - b2y1 + eps;
    float iw = fminf(b1x2, b2x2) - fmaxf(b1x1, b2x1); iw = fmaxf(iw, 0.f);
    float ih = fminf(b1y2, b2y2) - fmaxf(b1y1, b2y1); ih = fmaxf(ih, 0.f);
    float inter = iw * ih;
    float uni = w1 * h1 + w2 * h2 - inter + eps;
    float iou = inter / uni;
    float cw = fmaxf(b1x2, b2x2) - fminf(b1x1, b2x1);
    float ch = fmaxf(b1y2, b2y2) - fminf(b1y1, b2y1);
    float c2 = cw * cw + ch * ch + eps;
    float dx = b2x1 + b2x2 - b1x1 - b1x2;
    float dy = b2y1 + b2y2 - b1y1 - b1y2;
    float rho2 = (dx * dx + dy * dy) * 0.25f;
    const float PI2 = 9.869604401089358f; // pi^2
    float d = atanf(w2 / h2) - atanf(w1 / h1);
    float v = (4.0f / PI2) * d * d;
    float al = v / (v - iou + 1.0f + eps);
    return iou - (rho2 / c2 + v * al);
}

// -------------------- K1: preprocess targets --------------------
__global__ void k_prep(const float* __restrict__ tg, int ntg,
                       int* gl, float* gb, float* mg) {
    int t = threadIdx.x;
    for (int i = t; i < BB * MM; i += blockDim.x) {
        gl[i] = 0;
        gb[i * 4 + 0] = 0.f; gb[i * 4 + 1] = 0.f; gb[i * 4 + 2] = 0.f; gb[i * 4 + 3] = 0.f;
    }
    __syncthreads();
    for (int i = t; i < ntg; i += blockDim.x) {
        int bi = (int)tg[i * 6 + 0];
        int cnt = 0;
        for (int j = 0; j < ntg; ++j) cnt += ((int)tg[j * 6 + 0] < bi) ? 1 : 0;
        int pos = i - cnt;
        if (bi >= 0 && bi < BB && pos >= 0 && pos < MM) {
            float cx = tg[i * 6 + 2] * 640.f, cy = tg[i * 6 + 3] * 640.f;
            float ww = tg[i * 6 + 4] * 640.f, hh = tg[i * 6 + 5] * 640.f;
            int o = bi * MM + pos;
            gl[o] = (int)tg[i * 6 + 1];
            gb[o * 4 + 0] = cx - ww * 0.5f; gb[o * 4 + 1] = cy - hh * 0.5f;
            gb[o * 4 + 2] = cx + ww * 0.5f; gb[o * 4 + 3] = cy + hh * 0.5f;
        }
    }
    __syncthreads();
    for (int i = t; i < BB * MM; i += blockDim.x) {
        float s = gb[i * 4] + gb[i * 4 + 1] + gb[i * 4 + 2] + gb[i * 4 + 3];
        mg[i] = (s > 0.f) ? 1.f : 0.f;
    }
}

// -------------------- K2: DFL decode -> pred boxes (feature units) --------------------
__global__ void k_decode(const float* __restrict__ f0, const float* __restrict__ f1,
                         const float* __restrict__ f2, float* __restrict__ pb) {
    int i = blockIdx.x * blockDim.x + threadIdx.x;
    if (i >= BB * AA) return;
    int a = i % AA, b = i / AA;
    float ax, ay, st; anchor_info(a, ax, ay, st);
    float d[4];
#pragma unroll
    for (int g = 0; g < 4; ++g) {
        float x[REGM]; float mx = -1e30f;
#pragma unroll
        for (int k = 0; k < REGM; ++k) { x[k] = feat_at(f0, f1, f2, b, g * REGM + k, a); mx = fmaxf(mx, x[k]); }
        float se = 0.f, ke = 0.f;
#pragma unroll
        for (int k = 0; k < REGM; ++k) { float e = expf(x[k] - mx); se += e; ke += (float)k * e; }
        d[g] = ke / se;
    }
    pb[(size_t)i * 4 + 0] = ax - d[0];
    pb[(size_t)i * 4 + 1] = ay - d[1];
    pb[(size_t)i * 4 + 2] = ax + d[2];
    pb[(size_t)i * 4 + 3] = ay + d[3];
}

// -------------------- K3: pairwise overlaps + align metric --------------------
__global__ void k_pair(const float* __restrict__ f0, const float* __restrict__ f1,
                       const float* __restrict__ f2, const float* __restrict__ pb,
                       const int* __restrict__ gl, const float* __restrict__ gb,
                       const float* __restrict__ mg,
                       float* __restrict__ ov, float* __restrict__ alg) {
    long long i = (long long)blockIdx.x * blockDim.x + threadIdx.x;
    if (i >= (long long)BB * MM * AA) return;
    int a = (int)(i % AA); int r = (int)(i / AA); int b = r / MM;
    float o = 0.f, s = 0.f;
    if (mg[r] > 0.f) {
        float ax, ay, st; anchor_info(a, ax, ay, st);
        float axi = ax * st, ayi = ay * st;
        float g0 = gb[r * 4 + 0], g1 = gb[r * 4 + 1], g2 = gb[r * 4 + 2], g3 = gb[r * 4 + 3];
        float mn = fminf(fminf(axi - g0, ayi - g1), fminf(g2 - axi, g3 - ayi));
        if (mn > 1e-9f) {
            size_t pi = ((size_t)b * AA + a) * 4;
            float px1 = pb[pi + 0] * st, py1 = pb[pi + 1] * st;
            float px2 = pb[pi + 2] * st, py2 = pb[pi + 3] * st;
            float c = ciou_f(g0, g1, g2, g3, px1, py1, px2, py2);
            o = fmaxf(c, 0.f);
            int lbl = gl[r];
            float x = feat_at(f0, f1, f2, b, 4 * REGM + lbl, a);
            float sc = 1.f / (1.f + expf(-x));
            float o2 = o * o;
            s = sqrtf(sc) * (o2 * o2 * o2);
        }
    }
    ov[i] = o; alg[i] = s;
}

// -------------------- K4: per-gt top-10 -> mask_pos --------------------
__global__ void k_topk(const float* __restrict__ alg, const float* __restrict__ gb,
                       const float* __restrict__ mg, float* __restrict__ mp) {
    int r = blockIdx.x;            // (b,m)
    int t = threadIdx.x;
    __shared__ float s_al[AA];
    __shared__ unsigned char flag[AA];
    __shared__ float redv[256];
    __shared__ int   redi[256];
    __shared__ int   sel[TOPKK];
    size_t row = (size_t)r * AA;
    if (mg[r] <= 0.f) {
        for (int i = t; i < AA; i += 256) mp[row + i] = 0.f;
        return;
    }
    for (int i = t; i < AA; i += 256) { s_al[i] = alg[row + i]; flag[i] = 0; }
    __syncthreads();
    for (int round = 0; round < TOPKK; ++round) {
        float bv = -1.f; int bi = 0x7fffffff;
        for (int i = t; i < AA; i += 256) {
            if (!flag[i]) {
                float v = s_al[i];
                if (v > bv || (v == bv && i < bi)) { bv = v; bi = i; }
            }
        }
        redv[t] = bv; redi[t] = bi;
        __syncthreads();
        for (int stp = 128; stp > 0; stp >>= 1) {
            if (t < stp) {
                if (redv[t + stp] > redv[t] ||
                    (redv[t + stp] == redv[t] && redi[t + stp] < redi[t])) {
                    redv[t] = redv[t + stp]; redi[t] = redi[t + stp];
                }
            }
            __syncthreads();
        }
        if (t == 0) { sel[round] = redi[0]; flag[redi[0]] = 1; }
        __syncthreads();
    }
    for (int i = t; i < AA; i += 256) mp[row + i] = 0.f;
    __syncthreads();
    if (t < TOPKK) {
        int a = sel[t];
        float ax, ay, st; anchor_info(a, ax, ay, st);
        float axi = ax * st, ayi = ay * st;
        float g0 = gb[r * 4 + 0], g1 = gb[r * 4 + 1], g2 = gb[r * 4 + 2], g3 = gb[r * 4 + 3];
        float mn = fminf(fminf(axi - g0, ayi - g1), fminf(g2 - axi, g3 - ayi));
        mp[row + a] = (mn > 1e-9f) ? 1.f : 0.f;
    }
}

// -------------------- K5: fg = sum_m mask_pos  via WMMA (exact 0/1 math) ----------
// One wave computes fg for 16 anchors: D = A(16x32) x B(32x16), A = mask_pos rows
// (anchor-major, K = gt index, zero-padded past MM), B = all-ones (layout-proof).
__global__ void __launch_bounds__(32) k_fg_wmma(const float* __restrict__ mp,
                                                float* __restrict__ fg0) {
    const int tilesPerB = AA / 16;                 // 525
    int tile = blockIdx.x % tilesPerB;
    int b    = blockIdx.x / tilesPerB;
    int a0   = tile * 16;
    int lane = threadIdx.x & 31;
    int mrow = lane & 15;                          // anchor row within tile
    int hi   = (lane >= 16) ? 8 : 0;
    v16h aM;
#pragma unroll
    for (int e = 0; e < 16; ++e) {
        int v = e >> 1, lo = e & 1;
        int K = (v < 4) ? (2 * v + lo + hi) : (16 + 2 * (v - 4) + lo + hi);
        float val = (K < MM) ? mp[((size_t)b * MM + K) * AA + a0 + mrow] : 0.f;
        aM[e] = (_Float16)val;
    }
    v16h bM;
#pragma unroll
    for (int e = 0; e < 16; ++e) bM[e] = (_Float16)1.0f;
    v8f c = {};
    c = __builtin_amdgcn_wmma_f32_16x16x32_f16(false, aM, false, bM,
                                               (short)0, c, false, false);
    // D layout: VGPR r, lanes0-15 -> M=r, N=lane; lanes16-31 -> M=r+8.
    // Every column of D equals the row-sum, so lane 0 / lane 16 suffice.
    if (lane == 0) {
#pragma unroll
        for (int rr = 0; rr < 8; ++rr) fg0[(size_t)b * AA + a0 + rr] = c[rr];
    }
    if (lane == 16) {
#pragma unroll
        for (int rr = 0; rr < 8; ++rr) fg0[(size_t)b * AA + a0 + 8 + rr] = c[rr];
    }
}

// -------------------- K6: resolve multi-gt anchors --------------------
__global__ void k_resolve(const float* __restrict__ fg0, float* __restrict__ mp,
                          const float* __restrict__ ov,
                          float* __restrict__ fgb, int* __restrict__ ti) {
    int i = blockIdx.x * blockDim.x + threadIdx.x;
    if (i >= BB * AA) return;
    int a = i % AA, b = i / AA;
    float f0v = fg0[i];
    float bestov = -1.f; int mbest = 0; int first1 = -1;
    for (int m = 0; m < MM; ++m) {
        size_t k = ((size_t)b * MM + m) * AA + a;
        float ovv = ov[k];
        if (ovv > bestov) { bestov = ovv; mbest = m; }
        if (mp[k] > 0.5f && first1 < 0) first1 = m;
    }
    if (f0v > 1.5f) {
        for (int m = 0; m < MM; ++m) {
            size_t k = ((size_t)b * MM + m) * AA + a;
            mp[k] = (m == mbest) ? 1.f : 0.f;
        }
        fgb[i] = 1.f; ti[i] = mbest;
    } else {
        fgb[i] = (first1 >= 0) ? 1.f : 0.f;
        ti[i]  = (first1 >= 0) ? first1 : 0;
    }
}

// -------------------- K7: per-gt maxima of align_pos / overlap_pos ---------------
__global__ void k_gtmax(const float* __restrict__ alg, const float* __restrict__ ov,
                        const float* __restrict__ mp,
                        float* __restrict__ pa, float* __restrict__ po) {
    int r = blockIdx.x; int t = threadIdx.x;
    __shared__ float sa[256], so[256];
    float ma = 0.f, mo = 0.f;
    for (int i = t; i < AA; i += 256) {
        size_t k = (size_t)r * AA + i;
        float m_ = mp[k];
        ma = fmaxf(ma, alg[k] * m_);
        mo = fmaxf(mo, ov[k] * m_);
    }
    sa[t] = ma; so[t] = mo;
    __syncthreads();
    for (int s = 128; s > 0; s >>= 1) {
        if (t < s) { sa[t] = fmaxf(sa[t], sa[t + s]); so[t] = fmaxf(so[t], so[t + s]); }
        __syncthreads();
    }
    if (t == 0) { pa[r] = sa[0]; po[r] = so[0]; }
}

// -------------------- K8: norm -> weight (= fg*norm), partial weight sums --------
__global__ void k_norm(const float* __restrict__ alg, const float* __restrict__ mp,
                       const float* __restrict__ pa, const float* __restrict__ po,
                       const float* __restrict__ fgb,
                       float* __restrict__ wgt, float* __restrict__ pw) {
    int i = blockIdx.x * blockDim.x + threadIdx.x;
    int t = threadIdx.x;
    __shared__ float sm[256];
    float w = 0.f;
    if (i < BB * AA) {
        int a = i % AA, b = i / AA;
        float nm = 0.f;
        for (int m = 0; m < MM; ++m) {
            size_t k = ((size_t)b * MM + m) * AA + a;
            int r = b * MM + m;
            float v = alg[k] * mp[k] * po[r] / (pa[r] + 1e-9f);
            nm = fmaxf(nm, v);
        }
        w = fgb[i] * nm;
        wgt[i] = w;
    }
    sm[t] = w;
    __syncthreads();
    for (int s = 128; s > 0; s >>= 1) { if (t < s) sm[t] += sm[t + s]; __syncthreads(); }
    if (t == 0) pw[blockIdx.x] = sm[0];
}

// -------------------- K9: BCE classification partial sums --------------------
__global__ void k_cls(const float* __restrict__ f0, const float* __restrict__ f1,
                      const float* __restrict__ f2, const int* __restrict__ gl,
                      const int* __restrict__ ti, const float* __restrict__ wgt,
                      float* __restrict__ pc) {
    int i = blockIdx.x * blockDim.x + threadIdx.x;
    int t = threadIdx.x;
    __shared__ float sm[256];
    float acc = 0.f;
    if (i < BB * AA) {
        int a = i % AA, b = i / AA;
        for (int c = 0; c < NCLS; ++c) {
            float x = feat_at(f0, f1, f2, b, 4 * REGM + c, a);
            acc += fmaxf(x, 0.f) + log1pf(expf(-fabsf(x)));
        }
        int lbl = gl[b * MM + ti[i]];
        acc -= feat_at(f0, f1, f2, b, 4 * REGM + lbl, a) * wgt[i];
    }
    sm[t] = acc;
    __syncthreads();
    for (int s = 128; s > 0; s >>= 1) { if (t < s) sm[t] += sm[t + s]; __syncthreads(); }
    if (t == 0) pc[blockIdx.x] = sm[0];
}

// -------------------- K10: CIoU box + DFL partial sums --------------------
__global__ void k_box(const float* __restrict__ f0, const float* __restrict__ f1,
                      const float* __restrict__ f2, const float* __restrict__ pb,
                      const float* __restrict__ gb, const int* __restrict__ ti,
                      const float* __restrict__ fgb, const float* __restrict__ wgt,
                      float* __restrict__ pbox, float* __restrict__ pdfl) {
    int i = blockIdx.x * blockDim.x + threadIdx.x;
    int t = threadIdx.x;
    __shared__ float sb[256], sd[256];
    float accb = 0.f, accd = 0.f;
    if (i < BB * AA && fgb[i] > 0.f) {
        int a = i % AA, b = i / AA;
        float ax, ay, st; anchor_info(a, ax, ay, st);
        const float* g = &gb[(b * MM + ti[i]) * 4];
        float t0 = g[0] / st, t1 = g[1] / st, t2 = g[2] / st, t3 = g[3] / st;
        size_t pi = (size_t)i * 4;
        float px1 = pb[pi + 0], py1 = pb[pi + 1], px2 = pb[pi + 2], py2 = pb[pi + 3];
        float iou = ciou_f(px1, py1, px2, py2, t0, t1, t2, t3);
        float w = wgt[i];
        accb = (1.f - iou) * w;
        float tl4[4] = { ax - t0, ay - t1, t2 - ax, t3 - ay };
        float dsum = 0.f;
        for (int g4 = 0; g4 < 4; ++g4) {
            float tv = fminf(fmaxf(tl4[g4], 0.f), (float)(REGM - 1) - 0.01f);
            int tl = (int)tv;
            float wl = (float)(tl + 1) - tv;
            float xs[REGM]; float mx = -1e30f;
#pragma unroll
            for (int k = 0; k < REGM; ++k) { xs[k] = feat_at(f0, f1, f2, b, g4 * REGM + k, a); mx = fmaxf(mx, xs[k]); }
            float se = 0.f;
#pragma unroll
            for (int k = 0; k < REGM; ++k) se += expf(xs[k] - mx);
            float lse = logf(se);
            float lp_l = xs[tl] - mx - lse;
            float lp_r = xs[tl + 1] - mx - lse;
            dsum += (-lp_l) * wl + (-lp_r) * (1.f - wl);
        }
        accd = dsum * 0.25f * w;
    }
    sb[t] = accb; sd[t] = accd;
    __syncthreads();
    for (int s = 128; s > 0; s >>= 1) { if (t < s) { sb[t] += sb[t + s]; sd[t] += sd[t + s]; } __syncthreads(); }
    if (t == 0) { pbox[blockIdx.x] = sb[0]; pdfl[blockIdx.x] = sd[0]; }
}

// -------------------- K11: final combine (deterministic) --------------------
__global__ void k_final(const float* __restrict__ pw, const float* __restrict__ pc,
                        const float* __restrict__ pbox, const float* __restrict__ pdfl,
                        int nb, float* __restrict__ out) {
    int t = threadIdx.x;
    __shared__ float s0[256], s1[256], s2[256], s3[256];
    float w = 0.f, c = 0.f, bx = 0.f, dl = 0.f;
    for (int i = t; i < nb; i += 256) { w += pw[i]; c += pc[i]; bx += pbox[i]; dl += pdfl[i]; }
    s0[t] = w; s1[t] = c; s2[t] = bx; s3[t] = dl;
    __syncthreads();
    for (int s = 128; s > 0; s >>= 1) {
        if (t < s) { s0[t] += s0[t + s]; s1[t] += s1[t + s]; s2[t] += s2[t + s]; s3[t] += s3[t + s]; }
        __syncthreads();
    }
    if (t == 0) {
        float tss = fmaxf(s0[0], 1.0f);
        float lb = s2[0] / tss * 7.5f;
        float lc = s1[0] / tss * 0.5f;
        float ld = s3[0] / tss * 1.5f;
        out[0] = (lb + lc + ld) * (float)BB;
        out[1] = lb; out[2] = lc; out[3] = ld;
    }
}

// -------------------- launcher --------------------
extern "C" void kernel_launch(void* const* d_in, const int* in_sizes, int n_in,
                              void* d_out, int out_size, void* d_ws, size_t ws_size,
                              hipStream_t stream) {
    const float* f0 = (const float*)d_in[0];
    const float* f1 = (const float*)d_in[1];
    const float* f2 = (const float*)d_in[2];
    const float* tg = (const float*)d_in[3];
    int ntg = in_sizes[3] / 6;

    char* base = (char*)d_ws;
    size_t off = 0;
    auto alloc = [&](size_t bytes) -> void* {
        void* p = base + off;
        off += (bytes + 255) & ~(size_t)255;
        return p;
    };
    int*   gl   = (int*)  alloc((size_t)BB * MM * 4);
    float* gb   = (float*)alloc((size_t)BB * MM * 4 * 4);
    float* mg   = (float*)alloc((size_t)BB * MM * 4);
    float* pb   = (float*)alloc((size_t)BB * AA * 4 * 4);
    float* ov   = (float*)alloc((size_t)BB * MM * AA * 4);
    float* alg  = (float*)alloc((size_t)BB * MM * AA * 4);
    float* mp   = (float*)alloc((size_t)BB * MM * AA * 4);
    float* fg0  = (float*)alloc((size_t)BB * AA * 4);
    float* fgb  = (float*)alloc((size_t)BB * AA * 4);
    int*   ti   = (int*)  alloc((size_t)BB * AA * 4);
    float* wgt  = (float*)alloc((size_t)BB * AA * 4);
    float* pa   = (float*)alloc((size_t)BB * MM * 4);
    float* po   = (float*)alloc((size_t)BB * MM * 4);
    float* pw   = (float*)alloc((size_t)NBLK * 4);
    float* pc   = (float*)alloc((size_t)NBLK * 4);
    float* pbox = (float*)alloc((size_t)NBLK * 4);
    float* pdfl = (float*)alloc((size_t)NBLK * 4);

    k_prep   <<<1, 1024, 0, stream>>>(tg, ntg, gl, gb, mg);
    k_decode <<<NBLK, 256, 0, stream>>>(f0, f1, f2, pb);
    k_pair   <<<(BB * MM * AA) / 256, 256, 0, stream>>>(f0, f1, f2, pb, gl, gb, mg, ov, alg);
    k_topk   <<<BB * MM, 256, 0, stream>>>(alg, gb, mg, mp);
    k_fg_wmma<<<BB * (AA / 16), 32, 0, stream>>>(mp, fg0);
    k_resolve<<<NBLK, 256, 0, stream>>>(fg0, mp, ov, fgb, ti);
    k_gtmax  <<<BB * MM, 256, 0, stream>>>(alg, ov, mp, pa, po);
    k_norm   <<<NBLK, 256, 0, stream>>>(alg, mp, pa, po, fgb, wgt, pw);
    k_cls    <<<NBLK, 256, 0, stream>>>(f0, f1, f2, gl, ti, wgt, pc);
    k_box    <<<NBLK, 256, 0, stream>>>(f0, f1, f2, pb, gb, ti, fgb, wgt, pbox, pdfl);
    k_final  <<<1, 256, 0, stream>>>(pw, pc, pbox, pdfl, NBLK, (float*)d_out);
}
